// DGCNN_semseg_46514495816098
// MI455X (gfx1250) — compile-verified, matrix operand order checked by reading, not run on hardware
//
#include <hip/hip_runtime.h>

// ---------------------------------------------------------------------------
// DGCNN semseg forward for MI455X (gfx1250), wave32 + WMMA.
// B=8, N=4096, K=20, OUT=13.
// - GEMMs: v_wmma_f32_16x16x32_bf16 with pre-packed bf16 weights (B-frag
//   layout) and bf16 activation mirrors (wide vector loads).
// - KNN Gram matrices: v_wmma_f32_16x16x4_f32; top-20 kept as index-packed
//   f32 keys with a branchless min/max bubble insert.
// ---------------------------------------------------------------------------

typedef __bf16 bf16_t;
typedef __attribute__((ext_vector_type(16))) __bf16 v16bf;
typedef __attribute__((ext_vector_type(8)))  __bf16 v8bf;
typedef __attribute__((ext_vector_type(8)))  float  v8f;
typedef __attribute__((ext_vector_type(4)))  float  v4f;
typedef __attribute__((ext_vector_type(2)))  float  v2f;

#define BATCH 8
#define NPTS  4096
#define KNN_K 20
#define NROWS_EDGE 655360   // B*N*K
#define NROWS_MLP  32768    // B*N
#define LRELU(v) ((v) > 0.0f ? (v) : 0.2f * (v))

// ---------------- fragment helpers (ISA 7.12.2 layouts) --------------------

// A fragment halves are contiguous: elements j=0..7 at p[0..7], j=8..15 at
// p[16..23] (p already offset by (row)*ld + k0 + hi*8).  Two 16B loads.
__device__ __forceinline__ v16bf load_frag_rm(const bf16_t* p) {
  v8bf lo = *(const v8bf*)p;
  v8bf hh = *(const v8bf*)(p + 16);
  return __builtin_shufflevector(lo, hh, 0, 1, 2, 3, 4, 5, 6, 7, 8, 9, 10, 11,
                                 12, 13, 14, 15);
}

// B fragment from pre-packed weights: one aligned 32-byte load per lane.
__device__ __forceinline__ v16bf load_B_pk(const bf16_t* __restrict__ P,
                                           int CT, int kt, int ct, int lane) {
  return *(const v16bf*)(P + ((size_t)((kt * CT + ct) * 32 + lane) << 4));
}

__device__ __forceinline__ v8f wmma_bf16(v16bf a, v16bf b, v8f c) {
  return __builtin_amdgcn_wmma_f32_16x16x32_bf16(false, a, false, b, (short)0,
                                                 c, false, false);
}

// per-column sum / sum-of-squares reduction of two 16x16 accumulator tiles.
__device__ __forceinline__ void stats_red(const v8f& a0, const v8f& a1,
                                          float* sum, float* sq, int col,
                                          int lane) {
  float s = 0.0f, q = 0.0f;
#pragma unroll
  for (int i = 0; i < 8; ++i) {
    float v = a0[i]; s += v; q += v * v;
    v = a1[i];       s += v; q += v * v;
  }
  s += __shfl_xor(s, 16, 32);
  q += __shfl_xor(q, 16, 32);
  if ((lane >> 4) == 0) {
    atomicAdd(&sum[col], s);
    atomicAdd(&sq[col], q);
  }
}

__device__ __forceinline__ void atomicMaxFloat(float* addr, float val) {
  int cur = __float_as_int(*addr);
  while (val > __int_as_float(cur)) {
    int prev = atomicCAS((int*)addr, cur, __float_as_int(val));
    if (prev == cur) break;
    cur = prev;
  }
}

// -------------------- pack / convert utility kernels -----------------------
__global__ void pack_w_kernel(const float* __restrict__ W,
                              bf16_t* __restrict__ P, int Kr, int Cr, int KT,
                              int CT) {
  const int e = blockIdx.x * 256 + threadIdx.x;
  if (e >= KT * CT * 512) return;
  const int j = e & 15, lane = (e >> 4) & 31, t = e >> 9;
  const int ct = t % CT, kt = t / CT;
  const int hi = lane >> 4;
  const int k = kt * 32 + hi * 8 + ((j & 8) << 1) + (j & 7);
  const int c = ct * 16 + (lane & 15);
  const float v = (k < Kr && c < Cr) ? W[k * Cr + c] : 0.0f;
  P[e] = (bf16_t)v;
}

__global__ void cvt_bf_kernel(const float* __restrict__ s,
                              bf16_t* __restrict__ d, int n) {
  const int i = blockIdx.x * 256 + threadIdx.x;
  if (i < n) d[i] = (bf16_t)s[i];
}

// x9p[p*12 + f] = f<3 ? pos : f<9 ? xfeat : 0   (stride 12 keeps 8B align)
__global__ void pack_x9_kernel(const float* __restrict__ pos,
                               const float* __restrict__ xf,
                               float* __restrict__ x9) {
  const int e = blockIdx.x * 256 + threadIdx.x;
  if (e >= NROWS_MLP * 12) return;
  const int p = e / 12, f = e % 12;
  float v = 0.0f;
  if (f < 3) v = pos[p * 3 + f];
  else if (f < 9) v = xf[p * 6 + (f - 3)];
  x9[e] = v;
}

__global__ void sqnorm_kernel(const float* __restrict__ f,
                              float* __restrict__ sq, int ld) {
  const int p = blockIdx.x * 256 + threadIdx.x;
  if (p >= NROWS_MLP) return;
  float s = 0.0f;
  for (int i = 0; i < ld; ++i) {
    const float v = f[p * ld + i];
    s = fmaf(v, v, s);
  }
  sq[p] = s;
}

// ------------------- KNN top-20 via f32 WMMA Gram tiles --------------------
// Block = 4 waves; wave w owns queries n0+w*16..+15 (B operand, registers).
// Loop m in 16-tiles: Gram tile via KT x v_wmma_f32_16x16x4_f32.  Candidates
// are ranked as f32 keys with the index packed in the low 12 mantissa bits
// (d >= 1 after +1 bias, so float order == (dist, idx) lexicographic order);
// insert = branchless v_min/v_max bubble over 20 registers.  Lane L and L^16
// see complementary halves of the database; merged at the end through LDS.
template <int LD, int KT>
__global__ __launch_bounds__(128) void knn_wmma_kernel(
    const float* __restrict__ feat, const float* __restrict__ sqb,
    int* __restrict__ idx_out) {
  __shared__ float md[128 * KNN_K];
  const int b = blockIdx.y;
  const int lane = threadIdx.x & 31, wave = threadIdx.x >> 5;
  const int hi = lane >> 4, l16 = lane & 15;
  const int n = blockIdx.x * 64 + wave * 16 + l16;  // query (batch-local)
  const int pq = (b << 12) + n;

  // query fragments: loop-invariant, one aligned v2f per k-step
  v2f qf[KT];
#pragma unroll
  for (int kt = 0; kt < KT; ++kt)
    qf[kt] = *(const v2f*)(feat + (size_t)pq * LD + kt * 4 + 2 * hi);
  const float sqn = sqb[pq];

  float dk[KNN_K];
#pragma unroll
  for (int j = 0; j < KNN_K; ++j) dk[j] = 3.4e38f;

  const float* fb = feat + ((size_t)(b << 12)) * LD;
  const float* sb = sqb + (b << 12);
#pragma unroll 1
  for (int m0 = 0; m0 < NPTS; m0 += 16) {
    v8f g = {};
#pragma unroll
    for (int kt = 0; kt < KT; ++kt) {
      v2f af = *(const v2f*)(fb + (size_t)(m0 + l16) * LD + kt * 4 + 2 * hi);
      g = __builtin_amdgcn_wmma_f32_16x16x4_f32(false, af, false, qf[kt],
                                                (short)0, g, false, false);
    }
    const v4f s0 = *(const v4f*)(sb + m0 + 8 * hi);
    const v4f s1 = *(const v4f*)(sb + m0 + 8 * hi + 4);
#pragma unroll
    for (int i = 0; i < 8; ++i) {
      const float sm = (i < 4) ? s0[i & 3] : s1[i & 3];
      const float d = fmaxf(sqn + sm - 2.0f * g[i], 0.0f) + 1.0f;
      float key = __int_as_float((__float_as_int(d) & ~0xFFF) |
                                 (m0 + 8 * hi + i));
      if (key < dk[KNN_K - 1]) {
#pragma unroll
        for (int j = 0; j < KNN_K; ++j) {
          const float o = dk[j];
          dk[j] = fminf(o, key);
          key = fmaxf(o, key);
        }
      }
    }
  }

  // merge sorted key lists of lane pair (L, L+16) -> final top-20 indices
  const int base = threadIdx.x * KNN_K;
#pragma unroll
  for (int j = 0; j < KNN_K; ++j) md[base + j] = dk[j];
  __syncthreads();
  if (hi == 0) {
    const int pa = base, pb = base + 16 * KNN_K;
    int ia = 0, ib = 0;
    const int o = pq * KNN_K;
#pragma unroll
    for (int j = 0; j < KNN_K; ++j) {
      const float da = md[pa + ia], db = md[pb + ib];
      int sel;
      if (da <= db) { sel = __float_as_int(da) & 0xFFF; ++ia; }
      else          { sel = __float_as_int(db) & 0xFFF; ++ib; }
      idx_out[o + j] = sel;
    }
  }
}

// --------------------------- Edge conv -------------------------------------
// One workgroup (4 waves) per point; 32-row tile (20 real + pad), 64 cols.
// PASS 0: stats(h@Wa). PASS 1: stats(bnlrelu(h@Wa)@Wb). PASS 2: final+max.
template <int FIN, int LDF, int KA, bool HASB, int PASS>
__global__ __launch_bounds__(128) void edgeconv_kernel(
    const float* __restrict__ feat, const int* __restrict__ knn,
    const bf16_t* __restrict__ Pa, const bf16_t* __restrict__ Pb,
    const float* __restrict__ scA, const float* __restrict__ shA,
    const float* __restrict__ scB, const float* __restrict__ shB,
    float* __restrict__ stat_sum, float* __restrict__ stat_sq,
    float* __restrict__ xoutf, bf16_t* __restrict__ cat3, int cbase) {
  __shared__ __align__(32) bf16_t Hs[32 * KA];
  __shared__ __align__(32) bf16_t As[32 * 64];
  __shared__ int nb[KNN_K];
  const int p = blockIdx.x;  // b*4096 + n
  const int b = p >> 12;
  const int tid = threadIdx.x;
  if (tid < KNN_K) nb[tid] = knn[p * KNN_K + tid];
  __syncthreads();

  for (int e = tid; e < 32 * KA; e += 128) {
    const int row = e / KA, col = e % KA;
    float v = 0.0f;
    if (row < KNN_K && col < 2 * FIN) {
      if (col < FIN) {
        v = feat[(size_t)p * LDF + col];
      } else {
        const int pj = (b << 12) + nb[row];
        const int f = col - FIN;
        v = feat[(size_t)pj * LDF + f] - feat[(size_t)p * LDF + f];
      }
    }
    Hs[row * KA + col] = (bf16_t)v;
  }
  __syncthreads();

  const int lane = tid & 31, wave = tid >> 5;
  const int col = wave * 16 + (lane & 15);
  const int m = lane & 15, hi = lane >> 4;
  const int hi8 = hi * 8;

  v8f acc0 = {}, acc1 = {};
#pragma unroll
  for (int kt = 0; kt < KA / 32; ++kt) {
    v16bf bw = load_B_pk(Pa, 4, kt, wave, lane);
    acc0 = wmma_bf16(load_frag_rm(Hs + m * KA + kt * 32 + hi8), bw, acc0);
    acc1 = wmma_bf16(load_frag_rm(Hs + (16 + m) * KA + kt * 32 + hi8), bw, acc1);
  }
  if constexpr (PASS == 0) {
    stats_red(acc0, acc1, stat_sum, stat_sq, col, lane);
    return;
  }

  {  // BN-a + lrelu, zero padded rows (>=20)
    const float s = scA[col], t = shA[col];
#pragma unroll
    for (int i = 0; i < 8; ++i) {
      float v0 = LRELU(fmaf(s, acc0[i], t));
      float v1 = LRELU(fmaf(s, acc1[i], t));
      if (16 + i + 8 * hi >= KNN_K) v1 = 0.0f;
      acc0[i] = v0;
      acc1[i] = v1;
    }
  }

  if constexpr (!HASB) {  // single-layer conv final: max over neighbors
    float mx = -3.4e38f;
#pragma unroll
    for (int i = 0; i < 8; ++i) {
      mx = fmaxf(mx, acc0[i]);
      if (16 + i + 8 * hi < KNN_K) mx = fmaxf(mx, acc1[i]);
    }
    mx = fmaxf(mx, __shfl_xor(mx, 16, 32));
    if (hi == 0) {
      xoutf[p * 64 + col] = mx;
      cat3[p * 192 + cbase + col] = (bf16_t)mx;
    }
    return;
  }

#pragma unroll
  for (int i = 0; i < 8; ++i) {
    As[(i + 8 * hi) * 64 + col] = (bf16_t)acc0[i];
    As[(16 + i + 8 * hi) * 64 + col] = (bf16_t)acc1[i];
  }
  __syncthreads();

  v8f bc0 = {}, bc1 = {};
#pragma unroll
  for (int kt = 0; kt < 2; ++kt) {
    v16bf bw = load_B_pk(Pb, 4, kt, wave, lane);
    bc0 = wmma_bf16(load_frag_rm(As + m * 64 + kt * 32 + hi8), bw, bc0);
    bc1 = wmma_bf16(load_frag_rm(As + (16 + m) * 64 + kt * 32 + hi8), bw, bc1);
  }
  if constexpr (PASS == 1) {
    stats_red(bc0, bc1, stat_sum, stat_sq, col, lane);
    return;
  }
  {  // final: BN-b + lrelu + max over neighbors
    const float s = scB[col], t = shB[col];
    float mx = -3.4e38f;
#pragma unroll
    for (int i = 0; i < 8; ++i) {
      mx = fmaxf(mx, LRELU(fmaf(s, bc0[i], t)));
      if (16 + i + 8 * hi < KNN_K) mx = fmaxf(mx, LRELU(fmaf(s, bc1[i], t)));
    }
    mx = fmaxf(mx, __shfl_xor(mx, 16, 32));
    if (hi == 0) {
      xoutf[p * 64 + col] = mx;
      cat3[p * 192 + cbase + col] = (bf16_t)mx;
    }
  }
}

// ----------------------- MLP stats / max / chain ---------------------------
// MODE 0: A = cat3 (K=192).  MODE 1: A = [pooled | cat3] (K=1216).
template <int MODE, int CT>
__global__ __launch_bounds__(128) void mlp_stats_kernel(
    const bf16_t* __restrict__ cat3, const bf16_t* __restrict__ pooledb,
    const bf16_t* __restrict__ P, float* __restrict__ stat_sum,
    float* __restrict__ stat_sq) {
  const int r0 = blockIdx.x * 32;
  const int lane = threadIdx.x & 31, wave = threadIdx.x >> 5;
  const int ct = blockIdx.y * 4 + wave;
  const int col = ct * 16 + (lane & 15);
  const int m = lane & 15, hi8 = (lane >> 4) * 8;
  v8f a0 = {}, a1 = {};
  int koff = 0;
  if constexpr (MODE == 1) {
    const bf16_t* pb = pooledb + (r0 >> 12) * 1024 + hi8;
    for (int kt = 0; kt < 32; ++kt) {
      v16bf aw = load_frag_rm(pb + kt * 32);  // row-invariant (same batch)
      v16bf bw = load_B_pk(P, CT, kt, ct, lane);
      a0 = wmma_bf16(aw, bw, a0);
      a1 = wmma_bf16(aw, bw, a1);
    }
    koff = 32;
  }
#pragma unroll
  for (int kt = 0; kt < 6; ++kt) {
    v16bf bw = load_B_pk(P, CT, koff + kt, ct, lane);
    a0 = wmma_bf16(load_frag_rm(cat3 + (r0 + m) * 192 + kt * 32 + hi8), bw, a0);
    a1 = wmma_bf16(load_frag_rm(cat3 + (r0 + 16 + m) * 192 + kt * 32 + hi8), bw, a1);
  }
  stats_red(a0, a1, stat_sum, stat_sq, col, lane);
}

__global__ __launch_bounds__(128) void wm1_max_kernel(
    const bf16_t* __restrict__ cat3, const bf16_t* __restrict__ P,
    const float* __restrict__ sc, const float* __restrict__ sh,
    float* __restrict__ pooled) {
  const int r0 = blockIdx.x * 32;
  const int b = r0 >> 12;
  const int lane = threadIdx.x & 31, wave = threadIdx.x >> 5;
  const int ct = blockIdx.y * 4 + wave;
  const int col = ct * 16 + (lane & 15);
  const int m = lane & 15, hi8 = (lane >> 4) * 8;
  v8f a0 = {}, a1 = {};
#pragma unroll
  for (int kt = 0; kt < 6; ++kt) {
    v16bf bw = load_B_pk(P, 64, kt, ct, lane);
    a0 = wmma_bf16(load_frag_rm(cat3 + (r0 + m) * 192 + kt * 32 + hi8), bw, a0);
    a1 = wmma_bf16(load_frag_rm(cat3 + (r0 + 16 + m) * 192 + kt * 32 + hi8), bw, a1);
  }
  const float s = sc[col], t = sh[col];
  float mx = -3.4e38f;
#pragma unroll
  for (int i = 0; i < 8; ++i) {
    mx = fmaxf(mx, LRELU(fmaf(s, a0[i], t)));
    mx = fmaxf(mx, LRELU(fmaf(s, a1[i], t)));
  }
  mx = fmaxf(mx, __shfl_xor(mx, 16, 32));
  if ((lane >> 4) == 0) atomicMaxFloat(&pooled[b * 1024 + col], mx);
}

// a = bnlrelu([pooled|cat3]@Wa) staged in LDS (bf16), optionally continue.
// FINAL=false: stats of a@Wb.   FINAL=true: b = bnlrelu(a@Wb); out = b@Wc.
template <bool FINAL>
__global__ __launch_bounds__(128) void chain_kernel(
    const bf16_t* __restrict__ cat3, const bf16_t* __restrict__ pooledb,
    const bf16_t* __restrict__ Pa, const float* __restrict__ scA,
    const float* __restrict__ shA, const bf16_t* __restrict__ Pb,
    const float* __restrict__ scB, const float* __restrict__ shB,
    const bf16_t* __restrict__ Pc, float* __restrict__ stat_sum,
    float* __restrict__ stat_sq, float* __restrict__ out) {
  __shared__ __align__(32) bf16_t As[32 * 512];
  __shared__ __align__(32) bf16_t Bs[32 * 256];
  const int r0 = blockIdx.x * 32;
  const int lane = threadIdx.x & 31, wave = threadIdx.x >> 5;
  const int m = lane & 15, hi = lane >> 4, hi8 = hi * 8;

  // stage 1: a = bnlrelu(cat4 @ Wa)  (512 cols, 8 col-blocks of 64)
  for (int cb = 0; cb < 8; ++cb) {
    const int ct = cb * 4 + wave;
    const int col = ct * 16 + (lane & 15);
    v8f a0 = {}, a1 = {};
    {
      const bf16_t* pb = pooledb + (r0 >> 12) * 1024 + hi8;
      for (int kt = 0; kt < 32; ++kt) {
        v16bf aw = load_frag_rm(pb + kt * 32);
        v16bf bw = load_B_pk(Pa, 32, kt, ct, lane);
        a0 = wmma_bf16(aw, bw, a0);
        a1 = wmma_bf16(aw, bw, a1);
      }
    }
#pragma unroll
    for (int kt = 0; kt < 6; ++kt) {
      v16bf bw = load_B_pk(Pa, 32, 32 + kt, ct, lane);
      a0 = wmma_bf16(load_frag_rm(cat3 + (r0 + m) * 192 + kt * 32 + hi8), bw, a0);
      a1 = wmma_bf16(load_frag_rm(cat3 + (r0 + 16 + m) * 192 + kt * 32 + hi8), bw, a1);
    }
    const float s = scA[col], t = shA[col];
#pragma unroll
    for (int i = 0; i < 8; ++i) {
      As[(i + 8 * hi) * 512 + col] = (bf16_t)LRELU(fmaf(s, a0[i], t));
      As[(16 + i + 8 * hi) * 512 + col] = (bf16_t)LRELU(fmaf(s, a1[i], t));
    }
  }
  __syncthreads();

  // stage 2: a @ Wb (256 cols, 4 col-blocks)
  for (int cb = 0; cb < 4; ++cb) {
    const int ct = cb * 4 + wave;
    const int col = ct * 16 + (lane & 15);
    v8f b0 = {}, b1 = {};
#pragma unroll
    for (int kt = 0; kt < 16; ++kt) {
      v16bf bw = load_B_pk(Pb, 16, kt, ct, lane);
      b0 = wmma_bf16(load_frag_rm(As + m * 512 + kt * 32 + hi8), bw, b0);
      b1 = wmma_bf16(load_frag_rm(As + (16 + m) * 512 + kt * 32 + hi8), bw, b1);
    }
    if constexpr (!FINAL) {
      stats_red(b0, b1, stat_sum, stat_sq, col, lane);
    } else {
      const float s = scB[col], t = shB[col];
#pragma unroll
      for (int i = 0; i < 8; ++i) {
        Bs[(i + 8 * hi) * 256 + col] = (bf16_t)LRELU(fmaf(s, b0[i], t));
        Bs[(16 + i + 8 * hi) * 256 + col] = (bf16_t)LRELU(fmaf(s, b1[i], t));
      }
    }
  }
  if constexpr (FINAL) {
    __syncthreads();
    if (wave == 0) {  // wave-uniform: EXEC all-ones inside
      const int col = lane & 15;
      v8f o0 = {}, o1 = {};
#pragma unroll
      for (int kt = 0; kt < 8; ++kt) {
        v16bf bw = load_B_pk(Pc, 1, kt, 0, lane);
        o0 = wmma_bf16(load_frag_rm(Bs + m * 256 + kt * 32 + hi8), bw, o0);
        o1 = wmma_bf16(load_frag_rm(Bs + (16 + m) * 256 + kt * 32 + hi8), bw, o1);
      }
      if (col < 13) {
#pragma unroll
        for (int i = 0; i < 8; ++i) {
          out[(r0 + i + 8 * hi) * 13 + col] = o0[i];
          out[(r0 + 16 + i + 8 * hi) * 13 + col] = o1[i];
        }
      }
    }
  }
}

// ------------------------- small utility kernels ---------------------------
__global__ void bn_finalize_kernel(const float* __restrict__ sum,
                                   const float* __restrict__ sq,
                                   const float* __restrict__ g,
                                   const float* __restrict__ beta,
                                   float* __restrict__ sc,
                                   float* __restrict__ sh, int C,
                                   float inv_count) {
  const int c = blockIdx.x * 128 + threadIdx.x;
  if (c >= C) return;
  const float m = sum[c] * inv_count;
  const float v = sq[c] * inv_count - m * m;
  const float s = g[c] * rsqrtf(v + 1e-5f);
  sc[c] = s;
  sh[c] = beta[c] - m * s;
}

__global__ void fill_kernel(float* __restrict__ p, int n, float v) {
  const int i = blockIdx.x * 128 + threadIdx.x;
  if (i < n) p[i] = v;
}

// ------------------------------ launcher -----------------------------------
extern "C" void kernel_launch(void* const* d_in, const int* in_sizes, int n_in,
                              void* d_out, int out_size, void* d_ws,
                              size_t ws_size, hipStream_t stream) {
  (void)in_sizes; (void)n_in; (void)out_size; (void)ws_size;
  const float* pos  = (const float*)d_in[0];
  const float* xf   = (const float*)d_in[1];
  const float* W1a  = (const float*)d_in[2];
  const float* g1a  = (const float*)d_in[3];
  const float* b1a  = (const float*)d_in[4];
  const float* W1b  = (const float*)d_in[5];
  const float* g1b  = (const float*)d_in[6];
  const float* b1b  = (const float*)d_in[7];
  const float* W2a  = (const float*)d_in[8];
  const float* g2a  = (const float*)d_in[9];
  const float* b2a  = (const float*)d_in[10];
  const float* W2b  = (const float*)d_in[11];
  const float* g2b  = (const float*)d_in[12];
  const float* b2b  = (const float*)d_in[13];
  const float* W3a  = (const float*)d_in[14];
  const float* g3a  = (const float*)d_in[15];
  const float* b3a  = (const float*)d_in[16];
  const float* Wm1  = (const float*)d_in[17];
  const float* gm1  = (const float*)d_in[18];
  const float* bm1  = (const float*)d_in[19];
  const float* Wa   = (const float*)d_in[20];
  const float* ga   = (const float*)d_in[21];
  const float* ba   = (const float*)d_in[22];
  const float* Wb   = (const float*)d_in[23];
  const float* gb   = (const float*)d_in[24];
  const float* bbv  = (const float*)d_in[25];
  const float* Wc   = (const float*)d_in[26];
  float* out = (float*)d_out;

  size_t off = 0;
  auto take = [&](size_t bytes) -> void* {
    void* p = (char*)d_ws + off;
    off += (bytes + 255) & ~(size_t)255;
    return p;
  };
  int* idx1 = (int*)take(NROWS_EDGE * sizeof(int));
  int* idx2 = (int*)take(NROWS_EDGE * sizeof(int));
  int* idx3 = (int*)take(NROWS_EDGE * sizeof(int));
  float* x9p = (float*)take((size_t)NROWS_MLP * 12 * sizeof(float));
  float* x1 = (float*)take(NROWS_MLP * 64 * sizeof(float));
  float* x2 = (float*)take(NROWS_MLP * 64 * sizeof(float));
  float* x3 = (float*)take(NROWS_MLP * 64 * sizeof(float));
  float* sqb = (float*)take(NROWS_MLP * sizeof(float));
  bf16_t* cat3 = (bf16_t*)take((size_t)NROWS_MLP * 192 * sizeof(bf16_t));
  float*  pooled  = (float*)take(BATCH * 1024 * sizeof(float));
  bf16_t* pooledb = (bf16_t*)take(BATCH * 1024 * sizeof(bf16_t));
  float* ssum = (float*)take(1024 * sizeof(float));
  float* ssq  = (float*)take(1024 * sizeof(float));
  float* sc1a = (float*)take(64 * 4);  float* sh1a = (float*)take(64 * 4);
  float* sc1b = (float*)take(64 * 4);  float* sh1b = (float*)take(64 * 4);
  float* sc2a = (float*)take(64 * 4);  float* sh2a = (float*)take(64 * 4);
  float* sc2b = (float*)take(64 * 4);  float* sh2b = (float*)take(64 * 4);
  float* sc3a = (float*)take(64 * 4);  float* sh3a = (float*)take(64 * 4);
  float* scm1 = (float*)take(1024 * 4); float* shm1 = (float*)take(1024 * 4);
  float* scwa = (float*)take(512 * 4);  float* shwa = (float*)take(512 * 4);
  float* scwb = (float*)take(256 * 4);  float* shwb = (float*)take(256 * 4);
  // packed weights (bf16, B-fragment layout), sizes = KT*CT*512 elems
  bf16_t* P1a = (bf16_t*)take(1  * 4  * 512 * 2);
  bf16_t* P1b = (bf16_t*)take(2  * 4  * 512 * 2);
  bf16_t* P2a = (bf16_t*)take(4  * 4  * 512 * 2);
  bf16_t* P2b = (bf16_t*)take(2  * 4  * 512 * 2);
  bf16_t* P3a = (bf16_t*)take(4  * 4  * 512 * 2);
  bf16_t* Pm1 = (bf16_t*)take(6  * 64 * 512 * 2);
  bf16_t* Pwa = (bf16_t*)take(38 * 32 * 512 * 2);
  bf16_t* Pwb = (bf16_t*)take(16 * 16 * 512 * 2);
  bf16_t* Pwc = (bf16_t*)take(8  * 1  * 512 * 2);

  const dim3 blk(128);
  const float invE = 1.0f / (float)NROWS_EDGE;
  const float invM = 1.0f / (float)NROWS_MLP;
  auto packs = [](int kt, int ct) { return (kt * ct * 512 + 255) / 256; };

  // ---- pack weights + conv1 input ----
  pack_w_kernel<<<packs(1, 4), 256, 0, stream>>>(W1a, P1a, 18, 64, 1, 4);
  pack_w_kernel<<<packs(2, 4), 256, 0, stream>>>(W1b, P1b, 64, 64, 2, 4);
  pack_w_kernel<<<packs(4, 4), 256, 0, stream>>>(W2a, P2a, 128, 64, 4, 4);
  pack_w_kernel<<<packs(2, 4), 256, 0, stream>>>(W2b, P2b, 64, 64, 2, 4);
  pack_w_kernel<<<packs(4, 4), 256, 0, stream>>>(W3a, P3a, 128, 64, 4, 4);
  pack_w_kernel<<<packs(6, 64), 256, 0, stream>>>(Wm1, Pm1, 192, 1024, 6, 64);
  pack_w_kernel<<<packs(38, 32), 256, 0, stream>>>(Wa, Pwa, 1216, 512, 38, 32);
  pack_w_kernel<<<packs(16, 16), 256, 0, stream>>>(Wb, Pwb, 512, 256, 16, 16);
  pack_w_kernel<<<packs(8, 1), 256, 0, stream>>>(Wc, Pwc, 256, 13, 8, 1);
  pack_x9_kernel<<<(NROWS_MLP * 12 + 255) / 256, 256, 0, stream>>>(pos, xf, x9p);

  // ---- edge conv 1 (input dim 9, stride 12) ----
  sqnorm_kernel<<<NROWS_MLP / 256, 256, 0, stream>>>(x9p, sqb, 12);
  knn_wmma_kernel<12, 3><<<dim3(64, BATCH), blk, 0, stream>>>(x9p, sqb, idx1);
  hipMemsetAsync(ssum, 0, 64 * 4, stream);
  hipMemsetAsync(ssq, 0, 64 * 4, stream);
  edgeconv_kernel<9, 12, 32, true, 0><<<NROWS_MLP, blk, 0, stream>>>(
      x9p, idx1, P1a, P1b, nullptr, nullptr, nullptr, nullptr, ssum, ssq,
      nullptr, nullptr, 0);
  bn_finalize_kernel<<<1, 128, 0, stream>>>(ssum, ssq, g1a, b1a, sc1a, sh1a, 64, invE);
  hipMemsetAsync(ssum, 0, 64 * 4, stream);
  hipMemsetAsync(ssq, 0, 64 * 4, stream);
  edgeconv_kernel<9, 12, 32, true, 1><<<NROWS_MLP, blk, 0, stream>>>(
      x9p, idx1, P1a, P1b, sc1a, sh1a, nullptr, nullptr, ssum, ssq, nullptr,
      nullptr, 0);
  bn_finalize_kernel<<<1, 128, 0, stream>>>(ssum, ssq, g1b, b1b, sc1b, sh1b, 64, invE);
  edgeconv_kernel<9, 12, 32, true, 2><<<NROWS_MLP, blk, 0, stream>>>(
      x9p, idx1, P1a, P1b, sc1a, sh1a, sc1b, sh1b, nullptr, nullptr, x1, cat3, 0);

  // ---- edge conv 2 (input = x1) ----
  sqnorm_kernel<<<NROWS_MLP / 256, 256, 0, stream>>>(x1, sqb, 64);
  knn_wmma_kernel<64, 16><<<dim3(64, BATCH), blk, 0, stream>>>(x1, sqb, idx2);
  hipMemsetAsync(ssum, 0, 64 * 4, stream);
  hipMemsetAsync(ssq, 0, 64 * 4, stream);
  edgeconv_kernel<64, 64, 128, true, 0><<<NROWS_MLP, blk, 0, stream>>>(
      x1, idx2, P2a, P2b, nullptr, nullptr, nullptr, nullptr, ssum, ssq,
      nullptr, nullptr, 0);
  bn_finalize_kernel<<<1, 128, 0, stream>>>(ssum, ssq, g2a, b2a, sc2a, sh2a, 64, invE);
  hipMemsetAsync(ssum, 0, 64 * 4, stream);
  hipMemsetAsync(ssq, 0, 64 * 4, stream);
  edgeconv_kernel<64, 64, 128, true, 1><<<NROWS_MLP, blk, 0, stream>>>(
      x1, idx2, P2a, P2b, sc2a, sh2a, nullptr, nullptr, ssum, ssq, nullptr,
      nullptr, 0);
  bn_finalize_kernel<<<1, 128, 0, stream>>>(ssum, ssq, g2b, b2b, sc2b, sh2b, 64, invE);
  edgeconv_kernel<64, 64, 128, true, 2><<<NROWS_MLP, blk, 0, stream>>>(
      x1, idx2, P2a, P2b, sc2a, sh2a, sc2b, sh2b, nullptr, nullptr, x2, cat3, 64);

  // ---- edge conv 3 (input = x2, single layer) ----
  sqnorm_kernel<<<NROWS_MLP / 256, 256, 0, stream>>>(x2, sqb, 64);
  knn_wmma_kernel<64, 16><<<dim3(64, BATCH), blk, 0, stream>>>(x2, sqb, idx3);
  hipMemsetAsync(ssum, 0, 64 * 4, stream);
  hipMemsetAsync(ssq, 0, 64 * 4, stream);
  edgeconv_kernel<64, 64, 128, false, 0><<<NROWS_MLP, blk, 0, stream>>>(
      x2, idx3, P3a, nullptr, nullptr, nullptr, nullptr, nullptr, ssum, ssq,
      nullptr, nullptr, 0);
  bn_finalize_kernel<<<1, 128, 0, stream>>>(ssum, ssq, g3a, b3a, sc3a, sh3a, 64, invE);
  edgeconv_kernel<64, 64, 128, false, 2><<<NROWS_MLP, blk, 0, stream>>>(
      x2, idx3, P3a, nullptr, sc3a, sh3a, nullptr, nullptr, nullptr, nullptr,
      x3, cat3, 128);

  // ---- Wm1 (192 -> 1024) + global max pool ----
  hipMemsetAsync(ssum, 0, 1024 * 4, stream);
  hipMemsetAsync(ssq, 0, 1024 * 4, stream);
  mlp_stats_kernel<0, 64><<<dim3(1024, 16), blk, 0, stream>>>(cat3, nullptr,
                                                              Pm1, ssum, ssq);
  bn_finalize_kernel<<<8, 128, 0, stream>>>(ssum, ssq, gm1, bm1, scm1, shm1, 1024, invM);
  fill_kernel<<<64, 128, 0, stream>>>(pooled, BATCH * 1024, -3.4e38f);
  wm1_max_kernel<<<dim3(1024, 16), blk, 0, stream>>>(cat3, Pm1, scm1, shm1, pooled);
  cvt_bf_kernel<<<32, 256, 0, stream>>>(pooled, pooledb, BATCH * 1024);

  // ---- Wa (1216 -> 512) stats ----
  hipMemsetAsync(ssum, 0, 512 * 4, stream);
  hipMemsetAsync(ssq, 0, 512 * 4, stream);
  mlp_stats_kernel<1, 32><<<dim3(1024, 8), blk, 0, stream>>>(cat3, pooledb,
                                                             Pwa, ssum, ssq);
  bn_finalize_kernel<<<4, 128, 0, stream>>>(ssum, ssq, ga, ba, scwa, shwa, 512, invM);

  // ---- Wb (512 -> 256) stats via recompute chain ----
  hipMemsetAsync(ssum, 0, 256 * 4, stream);
  hipMemsetAsync(ssq, 0, 256 * 4, stream);
  chain_kernel<false><<<1024, blk, 0, stream>>>(cat3, pooledb, Pwa, scwa, shwa,
                                                Pwb, nullptr, nullptr, nullptr,
                                                ssum, ssq, nullptr);
  bn_finalize_kernel<<<2, 128, 0, stream>>>(ssum, ssq, gb, bbv, scwb, shwb, 256, invM);

  // ---- final tail: Wa -> Wb -> Wc ----
  chain_kernel<true><<<1024, blk, 0, stream>>>(cat3, pooledb, Pwa, scwa, shwa,
                                               Pwb, scwb, shwb, Pwc, nullptr,
                                               nullptr, out);
}
